// QuantizedLMHead_62732292326102
// MI455X (gfx1250) — matrix operand order repaired and checked by reading, not compiled, not run on hardware
//
#include <hip/hip_runtime.h>

typedef __bf16 bf16_t;
typedef __attribute__((ext_vector_type(16))) __bf16 v16bf;
typedef __attribute__((ext_vector_type(8)))  __bf16 v8bf;
typedef __attribute__((ext_vector_type(2)))  __bf16 v2bf;
typedef __attribute__((ext_vector_type(8)))  float   v8f;
typedef __attribute__((ext_vector_type(4)))  float   v4f;
typedef __attribute__((ext_vector_type(4)))  unsigned int u32x4;
typedef __attribute__((ext_vector_type(8)))  unsigned int u32x8;

#define D_DIM  2048
#define R_DIM  2048
#define V_DIM  32000
#define N_ROWS 4096
#define CBK    16
#define EPSQ   1e-8f

#define BM 64
#define BN 256             // 8 waves as 2(M) x 4(N); each wave: 32x64 output
#define KT 64              // two WMMA k-steps per staging round
#define LDA (KT + 8)       // 72 bf16 = 144 B padded rows (16B aligned)
#define LDB (KT + 8)
#define THREADS 256

__device__ __forceinline__ v16bf make_v16(v8bf lo, v8bf hi) {
  return __builtin_shufflevector(lo, hi, 0,1,2,3,4,5,6,7,8,9,10,11,12,13,14,15);
}
__device__ __forceinline__ unsigned int pack2(float a, float b) {
  v2bf p; p[0] = (bf16_t)a; p[1] = (bf16_t)b;     // v_cvt_pk_bf16_f32
  return __builtin_bit_cast(unsigned int, p);
}

// ---------------- fragment loads (documented wave32 WMMA layouts) ----------
__device__ __forceinline__ v16bf load_afrag(const bf16_t* Alds, int row, int kOff, int half) {
  const int ka = kOff + half * 8;
  v8bf lo = *(const v8bf*)&Alds[row * LDA + ka];
  v8bf hi = *(const v8bf*)&Alds[row * LDA + ka + 16];
  return make_v16(lo, hi);
}
__device__ __forceinline__ v16bf load_bfrag(const bf16_t* Blds, int col, int kOff, int half) {
  const int kb = kOff + half * 8;
  v8bf lo = *(const v8bf*)&Blds[col * LDB + kb];
  v8bf hi = *(const v8bf*)&Blds[col * LDB + kb + 16];
  return make_v16(lo, hi);
}

// TDM: one 2D descriptor moves a 64x64 bf16 tile global->LDS, with LDS
// padding 4 DWORDs every 32 DWORDs == our 72-halfword padded row stride.
__device__ __forceinline__ void tdm_load_A(unsigned long long gaddr, unsigned int ldsaddr) {
  u32x4 g0;
  g0[0] = 1u;                                              // count=1, user mode
  g0[1] = ldsaddr;                                         // lds_addr
  g0[2] = (unsigned int)gaddr;                             // global_addr[31:0]
  g0[3] = (unsigned int)((gaddr >> 32) & 0x01ffffffu) | (2u << 30);  // addr hi | type=2
  u32x8 g1;
  g1[0] = (1u << 16)          // data_size = 1 (2 bytes)
        | (1u << 20)          // pad_enable
        | (4u << 22)          // pad_interval: 32 DWORDs (=128B tile row)
        | (3u << 25);         // pad_amount: 4 DWORDs (=16B -> 144B stride)
  g1[1] = (unsigned int)R_DIM << 16;    // tensor_dim0[15:0]
  g1[2] = (unsigned int)N_ROWS << 16;   // tensor_dim0 hi | tensor_dim1[15:0]
  g1[3] = (unsigned int)KT << 16;       // tensor_dim1 hi | tile_dim0 = 64
  g1[4] = (unsigned int)BM;             // tile_dim1 = 64, tile_dim2 = 0
  g1[5] = (unsigned int)R_DIM;          // tensor_dim0_stride (low 32)
  g1[6] = 0u;
  g1[7] = 0u;
  asm volatile("tensor_load_to_lds %0, %1" :: "s"(g0), "s"(g1) : "memory");
}

// -------- Kernel 1: s = x @ U (bf16 WMMA), fused sorted-codebook quantize --
__global__ __launch_bounds__(THREADS) void qproj_kernel(
    const float* __restrict__ x,         // [N_ROWS, D_DIM]
    const float* __restrict__ U,         // [D_DIM, R_DIM]
    const float* __restrict__ lambda_x,  // [R_DIM]
    const float* __restrict__ scale_p,   // [1]
    const float* __restrict__ coeff,     // [R_DIM]
    const float* __restrict__ codebook,  // [CBK] sorted ascending
    bf16_t* __restrict__ Aout)           // [N_ROWS, R_DIM] bf16
{
  __shared__ bf16_t Alds[BM * LDA];
  __shared__ bf16_t Blds[BN * LDB];

  const int tid  = threadIdx.x;
  const int wid  = tid >> 5;
  const int lane = tid & 31;
  const int half = lane >> 4;
  const int lr   = lane & 15;
  const int waveM = (wid & 1) * 32;
  const int waveN = (wid >> 1) * 64;
  const int nBase = blockIdx.x * BN;     // over R

  float cb[CBK], mid[CBK - 1];
#pragma unroll
  for (int i = 0; i < CBK; ++i) cb[i] = codebook[i];
#pragma unroll
  for (int i = 0; i < CBK - 1; ++i) mid[i] = 0.5f * (cb[i] + cb[i + 1]);
  const float safe_scale = fmaxf(scale_p[0], EPSQ);

  const int mPer   = (N_ROWS / BM) / gridDim.y;
  const int mStart = blockIdx.y * mPer;

  const int am = tid >> 2;               // A: row
  const int ak = (tid & 3) * 16;         // A: 16 f32 per thread
  const int bn = (tid & 63) << 2;        // B: 4 of 256 columns
  const int bk = (tid >> 6) << 4;        // B: 16 consecutive k

  for (int mi = mStart; mi < mStart + mPer; ++mi) {
    const int mBase = mi * BM;
    v8f acc[2][4];
#pragma unroll
    for (int a = 0; a < 2; ++a)
#pragma unroll
      for (int b = 0; b < 4; ++b) { v8f z = {}; acc[a][b] = z; }

    for (int kBase = 0; kBase < D_DIM; kBase += KT) {
      __syncthreads();
      // stage A: x [BM][KT] f32 -> bf16 row-major
      {
        const float* src = x + (size_t)(mBase + am) * D_DIM + kBase + ak;
        v4f f0 = *(const v4f*)(src);
        v4f f1 = *(const v4f*)(src + 4);
        v4f f2 = *(const v4f*)(src + 8);
        v4f f3 = *(const v4f*)(src + 12);
        v8bf h0, h1;
#pragma unroll
        for (int j = 0; j < 4; ++j) { h0[j] = (bf16_t)f0[j]; h0[4+j] = (bf16_t)f1[j]; }
#pragma unroll
        for (int j = 0; j < 4; ++j) { h1[j] = (bf16_t)f2[j]; h1[4+j] = (bf16_t)f3[j]; }
        *(v8bf*)&Alds[am * LDA + ak]     = h0;
        *(v8bf*)&Alds[am * LDA + ak + 8] = h1;
      }
      // stage B: U [KT][BN] f32 -> transposed bf16 [BN][KT] (packed b64 stores)
#pragma unroll
      for (int g = 0; g < 2; ++g) {
        const int kg = bk + g * 8;
        v4f row[8];
#pragma unroll
        for (int kk = 0; kk < 8; ++kk)
          row[kk] = *(const v4f*)(U + (size_t)(kBase + kg + kk) * R_DIM + nBase + bn);
#pragma unroll
        for (int c = 0; c < 4; ++c) {
#pragma unroll
          for (int q = 0; q < 2; ++q) {
            uint2 u;
            u.x = pack2(row[4*q+0][c], row[4*q+1][c]);
            u.y = pack2(row[4*q+2][c], row[4*q+3][c]);
            *(uint2*)&Blds[(bn + c) * LDB + kg + 4 * q] = u;
          }
        }
      }
      {
        int kpf = kBase + KT; if (kpf >= D_DIM) kpf = 0;
        __builtin_prefetch(U + (size_t)(kpf + bk)     * R_DIM + nBase + bn, 0, 1);
        __builtin_prefetch(U + (size_t)(kpf + bk + 8) * R_DIM + nBase + bn, 0, 1);
      }
      __syncthreads();

#pragma unroll
      for (int s = 0; s < 2; ++s) {
        const int kOff = s * 32;
        v16bf afr[2], bfr[4];
#pragma unroll
        for (int mt = 0; mt < 2; ++mt) afr[mt] = load_afrag(Alds, waveM + mt*16 + lr, kOff, half);
#pragma unroll
        for (int nt = 0; nt < 4; ++nt) bfr[nt] = load_bfrag(Blds, waveN + nt*16 + lr, kOff, half);
#pragma unroll
        for (int mt = 0; mt < 2; ++mt)
#pragma unroll
          for (int nt = 0; nt < 4; ++nt)
            acc[mt][nt] = __builtin_amdgcn_wmma_f32_16x16x32_bf16(
                false, afr[mt], false, bfr[nt], (short)0, acc[mt][nt], false, false);
      }
    }

    // fused quantize: z = s/(lambda*scale); nearest sorted level; * scale*coeff
#pragma unroll
    for (int mt = 0; mt < 2; ++mt) {
#pragma unroll
      for (int nt = 0; nt < 4; ++nt) {
        const int n = nBase + waveN + nt * 16 + lr;
        const float lam = lambda_x[n];
        const float sl  = (fabsf(lam) < EPSQ) ? EPSQ : lam;
        const float osc = safe_scale * coeff[n];
        const float inv = 1.0f / (sl * safe_scale);
#pragma unroll
        for (int r = 0; r < 8; ++r) {
          const int m = mBase + waveM + mt * 16 + half * 8 + r;
          const float z = acc[mt][nt][r] * inv;
          float q = cb[0];
#pragma unroll
          for (int i = 0; i < CBK - 1; ++i) q = (z > mid[i]) ? cb[i + 1] : q;
          Aout[(size_t)m * R_DIM + n] = (bf16_t)(q * osc);
        }
      }
    }
  }
}

// -------- Kernel 2: logits = A(bf16) @ Z_w ; TDM double-buffered A ---------
__global__ __launch_bounds__(THREADS) void lmhead_kernel(
    const bf16_t* __restrict__ A,    // [N_ROWS, R_DIM] bf16
    const float*  __restrict__ Zw,   // [R_DIM, V_DIM] f32
    float* __restrict__ out)         // [N_ROWS, V_DIM] f32
{
  __shared__ bf16_t Alds[2][BM * LDA];   // ping-pong TDM destination
  __shared__ bf16_t Blds[BN * LDB];

  const int tid  = threadIdx.x;
  const int wid  = tid >> 5;
  const int lane = tid & 31;
  const int half = lane >> 4;
  const int lr   = lane & 15;
  const int waveM = (wid & 1) * 32;
  const int waveN = (wid >> 1) * 64;
  const int nBase = blockIdx.x * BN;     // over V

  const int mPer   = (N_ROWS / BM) / gridDim.y;
  const int mStart = blockIdx.y * mPer;

  const unsigned long long Abase = (unsigned long long)(uintptr_t)A;
  const unsigned int ldsA[2] = {
      (unsigned int)(uintptr_t)&Alds[0][0],
      (unsigned int)(uintptr_t)&Alds[1][0] };

  const int bn = (tid & 63) << 2;
  const int bk = (tid >> 6) << 4;
  const int nT = R_DIM / KT;             // 32 k-tiles

  for (int mi = mStart; mi < mStart + mPer; ++mi) {
    const int mBase = mi * BM;
    v8f acc[2][4];
#pragma unroll
    for (int a = 0; a < 2; ++a)
#pragma unroll
      for (int b = 0; b < 4; ++b) { v8f z = {}; acc[a][b] = z; }

    // prologue: TDM tile 0 -> buffer 0 (one descriptor per workgroup)
    if (wid == 0)
      tdm_load_A(Abase + 2ull * (size_t)mBase * R_DIM, ldsA[0]);

    for (int r = 0; r < nT; ++r) {
      const int kBase = r * KT;
      const bool more = (r + 1 < nT);
      __syncthreads();
      // TDM: start next A tile into the other buffer while we work on this one
      if (wid == 0 && more)
        tdm_load_A(Abase + 2ull * ((size_t)mBase * R_DIM + kBase + KT),
                   ldsA[(r + 1) & 1]);
      // stage B: Zw [KT][BN] f32 -> transposed bf16 [BN][KT]
#pragma unroll
      for (int g = 0; g < 2; ++g) {
        const int kg = bk + g * 8;
        v4f row[8];
#pragma unroll
        for (int kk = 0; kk < 8; ++kk)
          row[kk] = *(const v4f*)(Zw + (size_t)(kBase + kg + kk) * V_DIM + nBase + bn);
#pragma unroll
        for (int c = 0; c < 4; ++c) {
#pragma unroll
          for (int q = 0; q < 2; ++q) {
            uint2 u;
            u.x = pack2(row[4*q+0][c], row[4*q+1][c]);
            u.y = pack2(row[4*q+2][c], row[4*q+3][c]);
            *(uint2*)&Blds[(bn + c) * LDB + kg + 4 * q] = u;
          }
        }
      }
      {
        int kpf = kBase + KT; if (kpf >= R_DIM) kpf = 0;   // wraps to next mi's tile 0
        __builtin_prefetch(Zw + (size_t)(kpf + bk)     * V_DIM + nBase + bn, 0, 1);
        __builtin_prefetch(Zw + (size_t)(kpf + bk + 8) * V_DIM + nBase + bn, 0, 1);
      }
      // in-order TDM completion: allow the r+1 prefetch to stay in flight
      if (more) __builtin_amdgcn_s_wait_tensorcnt((short)1);
      else      __builtin_amdgcn_s_wait_tensorcnt((short)0);
      __syncthreads();

      const bf16_t* Acur = &Alds[r & 1][0];
#pragma unroll
      for (int s = 0; s < 2; ++s) {
        const int kOff = s * 32;
        v16bf afr[2], bfr[4];
#pragma unroll
        for (int mt = 0; mt < 2; ++mt) afr[mt] = load_afrag(Acur, waveM + mt*16 + lr, kOff, half);
#pragma unroll
        for (int nt = 0; nt < 4; ++nt) bfr[nt] = load_bfrag(Blds, waveN + nt*16 + lr, kOff, half);
#pragma unroll
        for (int mt = 0; mt < 2; ++mt)
#pragma unroll
          for (int nt = 0; nt < 4; ++nt)
            acc[mt][nt] = __builtin_amdgcn_wmma_f32_16x16x32_bf16(
                false, afr[mt], false, bfr[nt], (short)0, acc[mt][nt], false, false);
      }
    }

    // store f32 output (documented C/D layout)
#pragma unroll
    for (int mt = 0; mt < 2; ++mt) {
#pragma unroll
      for (int nt = 0; nt < 4; ++nt) {
        const int n = nBase + waveN + nt * 16 + lr;
#pragma unroll
        for (int r = 0; r < 8; ++r) {
          const int m = mBase + waveM + mt * 16 + half * 8 + r;
          out[(size_t)m * V_DIM + n] = acc[mt][nt][r];
        }
      }
    }
  }
}

extern "C" void kernel_launch(void* const* d_in, const int* in_sizes, int n_in,
                              void* d_out, int out_size, void* d_ws, size_t ws_size,
                              hipStream_t stream) {
  (void)in_sizes; (void)n_in; (void)out_size; (void)ws_size;
  const float* x        = (const float*)d_in[0];  // hidden_states
  const float* U        = (const float*)d_in[1];
  const float* lambda_x = (const float*)d_in[2];
  const float* scale_x  = (const float*)d_in[3];
  const float* coeff    = (const float*)d_in[4];
  const float* Zw       = (const float*)d_in[5];
  const float* codebook = (const float*)d_in[6];
  float*  out = (float*)d_out;
  bf16_t* Aq  = (bf16_t*)d_ws;   // [N_ROWS, R_DIM] bf16 = 16.8 MB

  dim3 blk(THREADS);
  // GEMM1 + quantize: 8 N-stripes x 32 M-splits = 256 workgroups
  qproj_kernel<<<dim3(R_DIM / BN, 32), blk, 0, stream>>>(
      x, U, lambda_x, scale_x, coeff, codebook, Aq);
  // GEMM2: 125 N-stripes x 4 M-splits = 500 workgroups; Z_w stripe L2-shared
  lmhead_kernel<<<dim3(V_DIM / BN, 4), blk, 0, stream>>>(Aq, Zw, out);
}
